// LstmDecoder_33414845563698
// MI455X (gfx1250) — matrix operand order repaired and checked by reading, not compiled, not run on hardware
//
#include <hip/hip_runtime.h>

// ---------------------------------------------------------------------------
// LSTM decoder for MI455X (gfx1250), wave32, WMMA bf16 16x16x32, f32 accum.
//   inputs [256,256] -> Linear+LeakyReLU -> h0 [256,256]
//   xg1 = h0 @ Wih1.T + (bih1+bhh1)          (pass-1 input is time-constant)
//   pass1: h,c recurrence only (acc init = xg1)          -> buf1 [B,T,256] bf16
//   pass2: fused x@Wih1.T + h@Whh1.T recurrence           -> buf2 [B,T,256] bf16
//   pass3: fused x@Wih2.T + h@Whh2.T recurrence (OUT=96)  -> d_out [B,T,96] f32
// ---------------------------------------------------------------------------

typedef __bf16 bf16;
typedef __attribute__((ext_vector_type(16))) __bf16 v16bf;
typedef __attribute__((ext_vector_type(8)))  float  v8f;

union FragBF { v16bf v; float4 q[2]; };

__device__ __forceinline__ v8f wmma_bf16(v16bf a, v16bf b, v8f c) {
  return __builtin_amdgcn_wmma_f32_16x16x32_bf16(false, a, false, b, (short)0, c,
                                                 false, false);
}

// A-matrix 16x32 bf16 fragment (ISA 7.12.2): lane m = l&15 holds row m.
// lanes 0-15: elems 0..7 = K 0..7,  elems 8..15 = K 16..23
// lanes16-31: elems 0..7 = K 8..15, elems 8..15 = K 24..31
__device__ __forceinline__ v16bf load_a_frag(const bf16* base, size_t rowStride, int k0) {
  const int l  = threadIdx.x & 31;
  const int m  = l & 15;
  const int kb = (l >> 4) * 8;
  const bf16* p = base + (size_t)m * rowStride + (size_t)(k0 + kb);
  FragBF u;
  u.q[0] = *(const float4*)p;        // K = k0+kb .. +7
  u.q[1] = *(const float4*)(p + 16); // K = k0+16+kb .. +7
  return u.v;
}

// B-matrix 32x16 bf16 fragment: B[k][n] = W[ncol0+n][k0+k] (row-major W[N][K]).
// lane: col n = l&15, elems 0..7 = K (l>>4)*8 ..+7, elems 8..15 = K 16+(l>>4)*8 ..+7.
__device__ __forceinline__ v16bf load_b_frag(const bf16* W, int ldw, int ncol0, int k0) {
  const int l  = threadIdx.x & 31;
  const int n  = l & 15;
  const int kr = (l >> 4) * 8;
  const bf16* p = W + (size_t)(ncol0 + n) * (size_t)ldw + (size_t)(k0 + kr);
  FragBF u;
  u.q[0] = *(const float4*)p;
  u.q[1] = *(const float4*)(p + 16);
  return u.v;
}

// gfx1250 has native v_tanh_f32; use it if the toolchain exposes it.
#if __has_builtin(__builtin_amdgcn_tanhf)
__device__ __forceinline__ float tanh_(float x) { return __builtin_amdgcn_tanhf(x); }
#else
__device__ __forceinline__ float tanh_(float x) { return 1.f - 2.f / (1.f + __expf(2.f * x)); }
#endif
__device__ __forceinline__ float sigmoid_(float x) { return 0.5f * tanh_(0.5f * x) + 0.5f; }

// --------------------------- small helper kernels ---------------------------

__global__ void cvt_f32_bf16_kernel(const float* __restrict__ s, bf16* __restrict__ d, int n) {
  int i = blockIdx.x * blockDim.x + threadIdx.x;
  if (i < n) d[i] = (bf16)s[i];
}

__global__ void bias_add_kernel(const float* __restrict__ a, const float* __restrict__ b,
                                float* __restrict__ o, int n) {
  int i = blockIdx.x * blockDim.x + threadIdx.x;
  if (i < n) o[i] = a[i] + b[i];
}

// --------------------------- dense GEMM (WMMA) ------------------------------
// C[M,N] = act(A[M,K] @ W[N,K].T + bias[N]); one 16x16 tile per wave.
// MODE 0: bf16 out with LeakyReLU(0.2). MODE 1: f32 out, no activation.
template <int MODE>
__global__ void gemm_bias_act_kernel(const bf16* __restrict__ A, const bf16* __restrict__ W,
                                     const float* __restrict__ bias, void* __restrict__ out,
                                     int M, int N, int K) {
  const int wave = threadIdx.x >> 5;
  const int lane = threadIdx.x & 31;
  const int tile = blockIdx.x * (blockDim.x >> 5) + wave;
  const int ntn  = N >> 4;
  if (tile >= (M >> 4) * ntn) return;
  const int tm = tile / ntn, tn = tile % ntn;
  const int nn = lane & 15, half = lane >> 4;

  const float bv = bias[tn * 16 + nn];
  v8f acc;
#pragma unroll
  for (int r = 0; r < 8; ++r) acc[r] = bv;

  const bf16* abase = A + (size_t)tm * 16 * K;
  for (int k0 = 0; k0 < K; k0 += 32) {
    v16bf a = load_a_frag(abase, (size_t)K, k0);
    v16bf b = load_b_frag(W, K, tn * 16, k0);
    acc = wmma_bf16(a, b, acc);
  }
#pragma unroll
  for (int r = 0; r < 8; ++r) {
    const int Mr = tm * 16 + r + half * 8;
    float v = acc[r];
    if (MODE == 0) {
      v = (v >= 0.f) ? v : 0.2f * v;
      ((bf16*)out)[(size_t)Mr * N + tn * 16 + nn] = (bf16)v;
    } else {
      ((float*)out)[(size_t)Mr * N + tn * 16 + nn] = v;
    }
  }
}

// --------------------------- persistent LSTM layer ---------------------------
// One WG per 16 batch rows; HH/16 waves, wave w owns hidden units [w*16, w*16+16).
// Per step: gates[16,4*HH] = (x_t@Wih.T) + h@Whh.T + init; c in regs, h via LDS.
template <int HH, int KIN, bool CONST_X, bool WRITE_F32>
__global__ void lstm_layer_kernel(const bf16* __restrict__ xprev,  // [B,T,KIN] (if !CONST_X)
                                  const float* __restrict__ xg,    // [B,4*HH]  (if CONST_X)
                                  const bf16* __restrict__ Wih,    // [4*HH,KIN]
                                  const bf16* __restrict__ Whh,    // [4*HH,HH]
                                  const float* __restrict__ bias,  // [4*HH] combined
                                  bf16* __restrict__ out_bf,       // [B,T,HH]
                                  float* __restrict__ out_f32,     // [B,T,HH] (final)
                                  int T) {
  const int lane  = threadIdx.x & 31;
  const int wave  = threadIdx.x >> 5;
  const int half  = lane >> 4;
  const int nn    = lane & 15;
  const int bRow0 = blockIdx.x * 16;
  const int hid0  = wave * 16;

  __shared__ bf16 hlds[16 * HH];
  for (int i = threadIdx.x; i < 16 * HH; i += blockDim.x) hlds[i] = (bf16)0.f;

  // Per-step accumulator init: xg tile (pass 1) or broadcast bias (pass 2/3).
  v8f init[4];
  if constexpr (CONST_X) {
#pragma unroll
    for (int q = 0; q < 4; ++q)
#pragma unroll
      for (int r = 0; r < 8; ++r) {
        const int Mr = r + half * 8;
        init[q][r] = xg[(size_t)(bRow0 + Mr) * (4 * HH) + q * HH + hid0 + nn];
      }
  } else {
#pragma unroll
    for (int q = 0; q < 4; ++q) {
      const float bv = bias[q * HH + hid0 + nn];
#pragma unroll
      for (int r = 0; r < 8; ++r) init[q][r] = bv;
    }
  }

  v8f c;
#pragma unroll
  for (int r = 0; r < 8; ++r) c[r] = 0.f;

  constexpr int XK = KIN / 32;
  v16bf xf[XK];
  const size_t xRow = (size_t)T * KIN;   // elements
  const bf16* xb0 = nullptr;
  if constexpr (!CONST_X) {
    xb0 = xprev + (size_t)bRow0 * xRow;
#pragma unroll
    for (int kk = 0; kk < XK; ++kk) xf[kk] = load_a_frag(xb0, xRow, kk * 32);  // t=0
  }

  __syncthreads();

  for (int t = 0; t < T; ++t) {
    v8f acc[4] = {init[0], init[1], init[2], init[3]};

    // Input projection x_t @ Wih.T (fused; fragments prefetched one step ahead)
    if constexpr (!CONST_X) {
#pragma unroll
      for (int kk = 0; kk < XK; ++kk) {
#pragma unroll
        for (int q = 0; q < 4; ++q)
          acc[q] = wmma_bf16(xf[kk], load_b_frag(Wih, KIN, q * HH + hid0, kk * 32), acc[q]);
      }
      // register double-buffer: fragments for t+1
      if (t + 1 < T) {
        const bf16* xbn = xb0 + (size_t)(t + 1) * KIN;
#pragma unroll
        for (int kk = 0; kk < XK; ++kk) xf[kk] = load_a_frag(xbn, xRow, kk * 32);
      }
      // L2->near-cache prefetch for the t+2 tile (16 rows x KIN*2 bytes):
      // 64 x 128B lines covered by 32 lanes x 2 prefetches (wave 0 only).
      if (wave == 0 && t + 2 < T) {
        const char* pf = (const char*)(xb0 + (size_t)(t + 2) * KIN);
#pragma unroll
        for (int j = 0; j < 2; ++j) {
          const int line = lane + j * 32;       // 0..63
          const int row  = line >> 2;           // 0..15
          const int colB = (line & 3) * 128;    // byte offset within 512B row
          __builtin_prefetch(pf + (size_t)row * (xRow * sizeof(bf16)) + colB, 0, 3);
        }
      }
    }

    // Recurrent projection h_{t-1} @ Whh.T (h tile from LDS)
#pragma unroll
    for (int k0 = 0; k0 < HH; k0 += 32) {
      v16bf a = load_a_frag(hlds, (size_t)HH, k0);
#pragma unroll
      for (int q = 0; q < 4; ++q)
        acc[q] = wmma_bf16(a, load_b_frag(Whh, HH, q * HH + hid0, k0), acc[q]);
    }

    __syncthreads();  // all waves done reading h_{t-1} from LDS

    // Gate nonlinearities + state update (i, f, g, o PyTorch order)
    float hv[8];
#pragma unroll
    for (int r = 0; r < 8; ++r) {
      const float gi = sigmoid_(acc[0][r]);
      const float gf = sigmoid_(acc[1][r]);
      const float gg = tanh_(acc[2][r]);
      const float go = sigmoid_(acc[3][r]);
      const float cc = gf * c[r] + gi * gg;
      c[r] = cc;
      hv[r] = go * tanh_(cc);
    }

#pragma unroll
    for (int r = 0; r < 8; ++r) {
      const int Mr = r + half * 8;
      const bf16 hb = (bf16)hv[r];
      hlds[Mr * HH + hid0 + nn] = hb;
      const size_t oidx = ((size_t)(bRow0 + Mr) * T + t) * HH + hid0 + nn;
      if constexpr (WRITE_F32) out_f32[oidx] = hv[r];
      else                     out_bf[oidx] = hb;
    }

    __syncthreads();  // h_t visible for next step
  }
}

// ------------------------------- host driver --------------------------------

extern "C" void kernel_launch(void* const* d_in, const int* in_sizes, int n_in,
                              void* d_out, int out_size, void* d_ws, size_t ws_size,
                              hipStream_t stream) {
  constexpr int B = 256, T = 1024, H = 256, OUT = 96, LAT = 256;

  const float* f_inputs = (const float*)d_in[0];   // [256,256]
  const float* f_linW   = (const float*)d_in[1];   // [256,256]
  const float* f_linb   = (const float*)d_in[2];   // [256]
  const float* f_Wih1   = (const float*)d_in[3];   // [1024,256]
  const float* f_Whh1   = (const float*)d_in[4];   // [1024,256]
  const float* f_bih1   = (const float*)d_in[5];   // [1024]
  const float* f_bhh1   = (const float*)d_in[6];   // [1024]
  const float* f_Wih2   = (const float*)d_in[7];   // [384,256]
  const float* f_Whh2   = (const float*)d_in[8];   // [384,96]
  const float* f_bih2   = (const float*)d_in[9];   // [384]
  const float* f_bhh2   = (const float*)d_in[10];  // [384]
  float* out = (float*)d_out;                      // [256,1024,96]

  // ---- workspace layout (bytes) ----
  char* ws = (char*)d_ws;
  size_t off = 0;
  auto alloc = [&](size_t bytes) { char* p = ws + off; off += (bytes + 255) & ~size_t(255); return p; };
  bf16*  in_bf   = (bf16*)alloc((size_t)B * LAT * 2);
  bf16*  Wlin_bf = (bf16*)alloc((size_t)H * LAT * 2);
  bf16*  Wih1_bf = (bf16*)alloc((size_t)4 * H * H * 2);
  bf16*  Whh1_bf = (bf16*)alloc((size_t)4 * H * H * 2);
  bf16*  Wih2_bf = (bf16*)alloc((size_t)4 * OUT * H * 2);
  bf16*  Whh2_bf = (bf16*)alloc((size_t)4 * OUT * OUT * 2);
  bf16*  h0_bf   = (bf16*)alloc((size_t)B * H * 2);
  float* bc1     = (float*)alloc((size_t)4 * H * 4);
  float* bc2     = (float*)alloc((size_t)4 * OUT * 4);
  float* xg1     = (float*)alloc((size_t)B * 4 * H * 4);
  bf16*  buf1    = (bf16*)alloc((size_t)B * T * H * 2);
  bf16*  buf2    = (bf16*)alloc((size_t)B * T * H * 2);
  (void)ws_size; (void)n_in; (void)in_sizes; (void)out_size;

  auto cvt = [&](const float* s, bf16* d, int n) {
    cvt_f32_bf16_kernel<<<(n + 255) / 256, 256, 0, stream>>>(s, d, n);
  };
  cvt(f_inputs, in_bf,   B * LAT);
  cvt(f_linW,   Wlin_bf, H * LAT);
  cvt(f_Wih1,   Wih1_bf, 4 * H * H);
  cvt(f_Whh1,   Whh1_bf, 4 * H * H);
  cvt(f_Wih2,   Wih2_bf, 4 * OUT * H);
  cvt(f_Whh2,   Whh2_bf, 4 * OUT * OUT);
  bias_add_kernel<<<(4 * H + 255) / 256, 256, 0, stream>>>(f_bih1, f_bhh1, bc1, 4 * H);
  bias_add_kernel<<<(4 * OUT + 255) / 256, 256, 0, stream>>>(f_bih2, f_bhh2, bc2, 4 * OUT);

  // h0 = LeakyReLU(inputs @ lin_W.T + lin_b) -> bf16 [256,256]
  {
    int tiles = (B / 16) * (H / 16);                  // 256 tiles, 8 waves/block
    gemm_bias_act_kernel<0><<<tiles / 8, 256, 0, stream>>>(in_bf, Wlin_bf, f_linb,
                                                           (void*)h0_bf, B, H, LAT);
  }
  // xg1 = h0 @ Wih1.T + (bih1+bhh1) -> f32 [256,1024]  (time-constant input proj)
  {
    int tiles = (B / 16) * (4 * H / 16);              // 1024 tiles
    gemm_bias_act_kernel<1><<<tiles / 8, 256, 0, stream>>>(h0_bf, Wih1_bf, bc1,
                                                           (void*)xg1, B, 4 * H, LAT);
  }

  // Pass 1: recurrence only (const input), 16 WGs x 16 waves
  lstm_layer_kernel<H, H, true, false><<<B / 16, (H / 16) * 32, 0, stream>>>(
      nullptr, xg1, nullptr, Whh1_bf, nullptr, buf1, nullptr, T);
  // Pass 2: fused x@Wih1 + h@Whh1
  lstm_layer_kernel<H, H, false, false><<<B / 16, (H / 16) * 32, 0, stream>>>(
      buf1, nullptr, Wih1_bf, Whh1_bf, bc1, buf2, nullptr, T);
  // Pass 3 (final, OUT=96): fused x@Wih2 + h@Whh2, f32 out to d_out
  lstm_layer_kernel<OUT, H, false, true><<<B / 16, (OUT / 16) * 32, 0, stream>>>(
      buf2, nullptr, Wih2_bf, Whh2_bf, bc2, nullptr, out, T);
}
